// AttentionLayer_23811298689854
// MI455X (gfx1250) — matrix operand order repaired and checked by reading, not compiled, not run on hardware
//
#include <hip/hip_runtime.h>
#include <hip/hip_bf16.h>

typedef __bf16 bf16_t;
typedef __attribute__((ext_vector_type(16))) __bf16 v16bf;
typedef __attribute__((ext_vector_type(8)))  __bf16 v8bf;
typedef __attribute__((ext_vector_type(4)))  __bf16 v4bf;
typedef __attribute__((ext_vector_type(8)))  float  v8f;

#define B_   2
#define S_   2048
#define HID_ 1024
#define NH_  16
#define HD_  64

#if __has_builtin(__builtin_amdgcn_global_load_async_to_lds_b128)
#define HAVE_ASYNC_LDS 1
typedef int v4i_ __attribute__((vector_size(16)));   // matches builtin param type
#endif

__device__ __forceinline__ v16bf cat8(v8bf lo, v8bf hi) {
  return __builtin_shufflevector(lo, hi, 0,1,2,3,4,5,6,7,8,9,10,11,12,13,14,15);
}

// ---------------------------------------------------------------------------
// Stage 1a: f32 -> bf16 conversion (vectorized x4), for activations
// ---------------------------------------------------------------------------
__global__ void cvt_bf16_kernel(const float* __restrict__ in,
                                bf16_t* __restrict__ out, int n4) {
  int i = blockIdx.x * blockDim.x + threadIdx.x;
  if (i < n4) {
    float4 f = ((const float4*)in)[i];
    v4bf o;
    o[0] = (bf16_t)f.x; o[1] = (bf16_t)f.y;
    o[2] = (bf16_t)f.z; o[3] = (bf16_t)f.w;
    ((v4bf*)out)[i] = o;
  }
}

// ---------------------------------------------------------------------------
// Stage 1b: f32 -> bf16 conversion with transpose, for weights.
//   out[n][k] = (bf16) in[k][n], 1024x1024. Tiled 32x32 via LDS.
// ---------------------------------------------------------------------------
__global__ __launch_bounds__(256)
void cvt_transpose_kernel(const float* __restrict__ in, bf16_t* __restrict__ out) {
  __shared__ bf16_t tile[32][33];
  const int k0 = blockIdx.y * 32, n0 = blockIdx.x * 32;
  const int x  = threadIdx.x & 31;
  const int y0 = (threadIdx.x >> 5) * 4;
#pragma unroll
  for (int yy = 0; yy < 4; yy++) {
    int y = y0 + yy;
    tile[y][x] = (bf16_t)in[(size_t)(k0 + y) * HID_ + n0 + x];
  }
  __syncthreads();
#pragma unroll
  for (int yy = 0; yy < 4; yy++) {
    int y = y0 + yy;
    out[(size_t)(n0 + y) * HID_ + k0 + x] = tile[x][y];
  }
}

// ---------------------------------------------------------------------------
// Stage 2: fused QKV projection GEMM.
//   Y[M=4096, N=1024] = X[M,1024] @ W[1024,1024] + bias
//   Weights pre-transposed to [n][k] so both LDS tiles stage as b128 copies
//   (async-to-LDS on CDNA5 when available). blockIdx.z selects Q/K/V.
//   Q,K written bf16 [B,H,S,D]; V written transposed [B,H,D,S]
//   (C-layout rows are consecutive s -> b128 stores).
//   Block: 256 threads (8 waves), tile 128x128, K-chunk 32.
// ---------------------------------------------------------------------------
__global__ __launch_bounds__(256)
void qkv_gemm_kernel(const bf16_t* __restrict__ Xf, const bf16_t* __restrict__ Xt,
                     const bf16_t* __restrict__ Wqt, const bf16_t* __restrict__ Wkt,
                     const bf16_t* __restrict__ Wvt,
                     const float* __restrict__ bq, const float* __restrict__ bk,
                     const float* __restrict__ bv,
                     bf16_t* __restrict__ Qo, bf16_t* __restrict__ Ko,
                     bf16_t* __restrict__ Vto) {
  const int which = blockIdx.z;
  const bf16_t* X    = (which == 0) ? Xf : Xt;
  const bf16_t* Wt   = (which == 0) ? Wqt : (which == 1) ? Wkt : Wvt;
  const float*  bias = (which == 0) ? bq : (which == 1) ? bk : bv;

  __shared__ __align__(16) bf16_t As[128][32];  // [m][k]
  __shared__ __align__(16) bf16_t Bs[128][32];  // [n][k]

  const int tid  = threadIdx.x;
  const int lane = tid & 31;
  const int wave = tid >> 5;
  const int hi   = lane >> 4;
  const int lr   = lane & 15;
  const int mblk = blockIdx.y * 128;
  const int nblk = blockIdx.x * 128;
  const int wm   = (wave >> 2) * 64;
  const int wn   = (wave & 3) * 32;

  const int srow = tid >> 1;
  const int scol = (tid & 1) * 16;
  const bf16_t* aptr = X  + (size_t)(mblk + srow) * HID_ + scol;
  const bf16_t* bptr = Wt + (size_t)(nblk + srow) * HID_ + scol;

  const v8f vzero = {};
  v8f acc[4][2];
#pragma unroll
  for (int i = 0; i < 4; i++)
#pragma unroll
    for (int j = 0; j < 2; j++) acc[i][j] = vzero;

  for (int k0 = 0; k0 < HID_; k0 += 32) {
#ifdef HAVE_ASYNC_LDS
    __builtin_amdgcn_global_load_async_to_lds_b128(
        (v4i_*)(aptr + k0),     (v4i_*)&As[srow][scol],     0, 0);
    __builtin_amdgcn_global_load_async_to_lds_b128(
        (v4i_*)(aptr + k0 + 8), (v4i_*)&As[srow][scol + 8], 0, 0);
    __builtin_amdgcn_global_load_async_to_lds_b128(
        (v4i_*)(bptr + k0),     (v4i_*)&Bs[srow][scol],     0, 0);
    __builtin_amdgcn_global_load_async_to_lds_b128(
        (v4i_*)(bptr + k0 + 8), (v4i_*)&Bs[srow][scol + 8], 0, 0);
    asm volatile("s_wait_asynccnt 0x0" ::: "memory");
#else
    *(v8bf*)&As[srow][scol]     = *(const v8bf*)(aptr + k0);
    *(v8bf*)&As[srow][scol + 8] = *(const v8bf*)(aptr + k0 + 8);
    *(v8bf*)&Bs[srow][scol]     = *(const v8bf*)(bptr + k0);
    *(v8bf*)&Bs[srow][scol + 8] = *(const v8bf*)(bptr + k0 + 8);
#endif
    __builtin_prefetch(aptr + k0 + 32, 0, 1);
    __builtin_prefetch(bptr + k0 + 32, 0, 1);
    __syncthreads();

    const int akb = hi * 16;   // A: K elems {akb..+15}
    const int bkb = hi * 16;   // B: K elems {bkb..+15}
    v16bf afrag[4], bfrag[2];
#pragma unroll
    for (int i = 0; i < 4; i++) {
      const bf16_t* p = &As[wm + i * 16 + lr][0];
      afrag[i] = cat8(*(const v8bf*)(p + akb), *(const v8bf*)(p + akb + 8));
    }
#pragma unroll
    for (int j = 0; j < 2; j++) {
      const bf16_t* p = &Bs[wn + j * 16 + lr][0];
      bfrag[j] = cat8(*(const v8bf*)(p + bkb), *(const v8bf*)(p + bkb + 8));
    }
#pragma unroll
    for (int i = 0; i < 4; i++)
#pragma unroll
      for (int j = 0; j < 2; j++)
        acc[i][j] = __builtin_amdgcn_wmma_f32_16x16x32_bf16(
            false, afrag[i], false, bfrag[j], (short)0, acc[i][j], false, false);
    __syncthreads();
  }

  if (which != 2) {
    bf16_t* Out = (which == 0) ? Qo : Ko;   // [B,H,S,D]
#pragma unroll
    for (int i = 0; i < 4; i++) {
#pragma unroll
      for (int j = 0; j < 2; j++) {
        int n = nblk + wn + j * 16 + lr;
        float bval = bias[n];
        int h = n >> 6, d = n & 63;
#pragma unroll
        for (int r = 0; r < 8; r++) {
          int m = mblk + wm + i * 16 + r + hi * 8;
          int bb = m >> 11, s = m & 2047;
          Out[((((size_t)bb * NH_ + h) * S_) + s) * HD_ + d] =
              (bf16_t)(acc[i][j][r] + bval);
        }
      }
    }
  } else {                                   // V -> [B,H,D,S] via b128 stores
#pragma unroll
    for (int i = 0; i < 4; i++) {
#pragma unroll
      for (int j = 0; j < 2; j++) {
        int n = nblk + wn + j * 16 + lr;
        float bval = bias[n];
        int h = n >> 6, d = n & 63;
        int m0 = mblk + wm + i * 16 + hi * 8;  // 8 consecutive s values
        int bb = m0 >> 11, s0 = m0 & 2047;
        v8bf ov;
#pragma unroll
        for (int r = 0; r < 8; r++) ov[r] = (bf16_t)(acc[i][j][r] + bval);
        *(v8bf*)&Vto[(((size_t)bb * NH_ + h) * HD_ + d) * S_ + s0] = ov;
      }
    }
  }
}

// ---------------------------------------------------------------------------
// Stage 3: flash attention. grid = (S/64 f-tiles, B*H). 128 threads (4 waves),
// each wave owns 16 query rows; t processed in chunks of 64.
// K read [B,H,S,D] (natural B-layout for Q@K^T), V read [B,H,D,S]
// (natural B-layout for P@V). Only P round-trips through LDS.
// ---------------------------------------------------------------------------
__global__ __launch_bounds__(128)
void flash_attn_kernel(const bf16_t* __restrict__ Q, const bf16_t* __restrict__ K,
                       const bf16_t* __restrict__ Vt, const int* __restrict__ mask,
                       float* __restrict__ Out) {
  __shared__ __align__(16) bf16_t Ps[4][16][64];   // per-wave P [m][t]

  const int tid  = threadIdx.x;
  const int lane = tid & 31;
  const int wave = tid >> 5;
  const int hi   = lane >> 4;
  const int lr   = lane & 15;
  const int bh = blockIdx.y;
  const int b = bh >> 4, h = bh & 15;
  const int f0 = blockIdx.x * 64;
  const int fw = f0 + wave * 16;

  const bf16_t* Qbh = Q  + (((size_t)b * NH_ + h) * S_) * HD_;
  const bf16_t* Kbh = K  + (((size_t)b * NH_ + h) * S_) * HD_;
  const bf16_t* Vbh = Vt + (((size_t)b * NH_ + h) * HD_) * S_;

  // Q A-fragments (d = 0..31, 32..63)
  v16bf qa[2];
  {
    const bf16_t* qrow = Qbh + (size_t)(fw + lr) * HD_;
    const int kb = hi * 16;
#pragma unroll
    for (int c = 0; c < 2; c++)
      qa[c] = cat8(*(const v8bf*)(qrow + c * 32 + kb),
                   *(const v8bf*)(qrow + c * 32 + kb + 8));
  }

  float mrun[8], lrun[8];
  const v8f vzero = {};
  v8f acc[4];
#pragma unroll
  for (int r = 0; r < 8; r++) { mrun[r] = -1e30f; lrun[r] = 0.0f; }
#pragma unroll
  for (int j = 0; j < 4; j++) acc[j] = vzero;

  for (int t0 = 0; t0 < S_; t0 += 64) {
    if (t0 + 64 < S_) {
      __builtin_prefetch(Kbh + (size_t)(t0 + 64 + lane) * HD_, 0, 1);
      __builtin_prefetch(Vbh + (size_t)lane * S_ + t0 + 64, 0, 1);
      __builtin_prefetch(Vbh + (size_t)(lane + 32) * S_ + t0 + 64, 0, 1);
    }

    // --- scores: S[16 x 64] = Q @ K^T, B-frags straight from K rows ---
    v8f sc[4];
#pragma unroll
    for (int j = 0; j < 4; j++) {
      v8f s = vzero;
      const bf16_t* krow = Kbh + (size_t)(t0 + j * 16 + lr) * HD_;
#pragma unroll
      for (int c = 0; c < 2; c++) {
        v16bf kb_ = cat8(*(const v8bf*)(krow + c * 32 + hi * 16),
                         *(const v8bf*)(krow + c * 32 + hi * 16 + 8));
        s = __builtin_amdgcn_wmma_f32_16x16x32_bf16(
            false, qa[c], false, kb_, (short)0, s, false, false);
      }
      sc[j] = s;
    }

    // --- scale + additive mask ---
#pragma unroll
    for (int j = 0; j < 4; j++) {
      int tcol = t0 + j * 16 + lr;
#pragma unroll
      for (int r = 0; r < 8; r++) {
        int fm = fw + r + hi * 8;
        int mv = mask[((size_t)b * S_ + fm) * S_ + tcol];
        sc[j][r] = sc[j][r] * 0.125f + (1.0f - (float)mv) * -10000.0f;
      }
    }

    // --- online softmax (row reductions across the 16-lane half) ---
    float alpha[8];
#pragma unroll
    for (int r = 0; r < 8; r++) {
      float rm = sc[0][r];
#pragma unroll
      for (int j = 1; j < 4; j++) rm = fmaxf(rm, sc[j][r]);
#pragma unroll
      for (int x = 1; x < 16; x <<= 1) rm = fmaxf(rm, __shfl_xor(rm, x, 32));
      float mnew = fmaxf(mrun[r], rm);
      alpha[r] = __expf(mrun[r] - mnew);
      mrun[r]  = mnew;
      float rs = 0.0f;
#pragma unroll
      for (int j = 0; j < 4; j++) {
        float p = __expf(sc[j][r] - mnew);
        sc[j][r] = p;
        rs += p;
      }
#pragma unroll
      for (int x = 1; x < 16; x <<= 1) rs += __shfl_xor(rs, x, 32);
      lrun[r] = lrun[r] * alpha[r] + rs;
    }

    // --- rescale acc, spill P (C-layout) to LDS for A-layout reload ---
#pragma unroll
    for (int j = 0; j < 4; j++) {
#pragma unroll
      for (int r = 0; r < 8; r++) {
        acc[j][r] *= alpha[r];
        Ps[wave][r + hi * 8][j * 16 + lr] = (bf16_t)sc[j][r];
      }
    }
    __syncthreads();

    // --- ctx += P @ V (V B-frags straight from transposed V rows) ---
#pragma unroll
    for (int c = 0; c < 2; c++) {
      const bf16_t* prow = &Ps[wave][lr][c * 32];
      v16bf pa = cat8(*(const v8bf*)(prow + hi * 16),
                      *(const v8bf*)(prow + hi * 16 + 8));
#pragma unroll
      for (int j = 0; j < 4; j++) {
        const bf16_t* vrow = Vbh + (size_t)(j * 16 + lr) * S_ + t0 + c * 32 + hi * 16;
        v16bf vb_ = cat8(*(const v8bf*)vrow, *(const v8bf*)(vrow + 8));
        acc[j] = __builtin_amdgcn_wmma_f32_16x16x32_bf16(
            false, pa, false, vb_, (short)0, acc[j], false, false);
      }
    }
    __syncthreads();   // WAR guard before next chunk's P spill
  }

  // --- epilogue: normalize, merge heads -> [B, F, HID] f32 ---
#pragma unroll
  for (int j = 0; j < 4; j++) {
    int d = j * 16 + lr;
#pragma unroll
    for (int r = 0; r < 8; r++) {
      int fm = fw + r + hi * 8;
      float inv = 1.0f / lrun[r];
      Out[((size_t)b * S_ + fm) * HID_ + h * HD_ + d] = acc[j][r] * inv;
    }
  }
}

// ---------------------------------------------------------------------------
extern "C" void kernel_launch(void* const* d_in, const int* in_sizes, int n_in,
                              void* d_out, int out_size, void* d_ws, size_t ws_size,
                              hipStream_t stream) {
  const float* from_t = (const float*)d_in[0];
  const float* to_t   = (const float*)d_in[1];
  const int*   mask   = (const int*)d_in[2];
  const float* Wq = (const float*)d_in[3];
  const float* bq = (const float*)d_in[4];
  const float* Wk = (const float*)d_in[5];
  const float* bk = (const float*)d_in[6];
  const float* Wv = (const float*)d_in[7];
  const float* bv = (const float*)d_in[8];
  float* out = (float*)d_out;

  const size_t NX = (size_t)B_ * S_ * HID_;   // 4,194,304
  const size_t NW = (size_t)HID_ * HID_;      // 1,048,576

  char* ws = (char*)d_ws;
  bf16_t* Xf  = (bf16_t*)ws; ws += NX * 2;
  bf16_t* Xt  = (bf16_t*)ws; ws += NX * 2;
  bf16_t* Wqt = (bf16_t*)ws; ws += NW * 2;
  bf16_t* Wkt = (bf16_t*)ws; ws += NW * 2;
  bf16_t* Wvt = (bf16_t*)ws; ws += NW * 2;
  bf16_t* Qo  = (bf16_t*)ws; ws += NX * 2;
  bf16_t* Ko  = (bf16_t*)ws; ws += NX * 2;
  bf16_t* Vto = (bf16_t*)ws; ws += NX * 2;

  // Stage 1: conversions (+ weight transpose)
  {
    int nx4 = (int)(NX / 4);
    cvt_bf16_kernel<<<(nx4 + 255) / 256, 256, 0, stream>>>(from_t, Xf, nx4);
    cvt_bf16_kernel<<<(nx4 + 255) / 256, 256, 0, stream>>>(to_t,   Xt, nx4);
    dim3 tg(HID_ / 32, HID_ / 32);
    cvt_transpose_kernel<<<tg, 256, 0, stream>>>(Wq, Wqt);
    cvt_transpose_kernel<<<tg, 256, 0, stream>>>(Wk, Wkt);
    cvt_transpose_kernel<<<tg, 256, 0, stream>>>(Wv, Wvt);
  }

  // Stage 2: fused QKV GEMMs (z = 0:Q, 1:K, 2:V-transposed)
  {
    dim3 grid(HID_ / 128, (B_ * S_) / 128, 3);
    qkv_gemm_kernel<<<grid, 256, 0, stream>>>(Xf, Xt, Wqt, Wkt, Wvt,
                                              bq, bk, bv, Qo, Ko, Vto);
  }

  // Stage 3: flash attention
  {
    dim3 grid(S_ / 64, B_ * NH_);
    flash_attn_kernel<<<grid, 128, 0, stream>>>(Qo, Ko, Vto, mask, out);
  }
}
